// soft_module_19774029430842
// MI455X (gfx1250) — matrix operand order repaired and checked by reading, not compiled, not run on hardware
//
#include <hip/hip_runtime.h>
#include <hip/hip_bf16.h>

// MI455X (gfx1250) fused soft-MoE forward.
// Roofline: ~306 MB streamed (x=268MB, c1/c2/c3=38MB) @ 23.3 TB/s ~= 13 us;
// 2.2e10 FLOP done as v_wmma_f32_16x16x32_bf16 (80 per 16-row wave tile) is
// negligible against that. Weights live in LDS as bf16; inter-stage tiles are
// staged per-wave in LDS in plain h order (k = 16*half + s) so every WMMA operand is
// a contiguous pair of ds_load_b128.

typedef __attribute__((ext_vector_type(16))) __bf16   v16bf;
typedef __attribute__((ext_vector_type(8)))  float    v8f;
typedef __attribute__((ext_vector_type(8)))  unsigned v8u;

#define IN_DIM 256
#define HID    32
#define NUM    4
#define ROWS_PER_BLOCK 128
#define THREADS 256

// ---- LDS layout (bytes) ----
#define OFF_W1   0          // 16384 u32 (W1 bf16 pairs, [col=128][i=256])
#define OFF_W2   65536      // 2048 u32  (W2 bf16 pairs, [n][k][h])
#define OFF_W3   73728      // 2048 u32
#define OFF_TILE 81920      // 8 waves * 16*128 bf16 (u16) = 32768 B
#define OFF_C1   114688     // 2048 f32 (c1 rows of this block)
#define OFF_C2   122880     // 2048 f32
#define OFF_C3   131072     // 512 f32
#define OFF_B1   133120     // 128 f32
#define OFF_B2   133632     // 128 f32
#define OFF_B3   134144     // 128 f32
#define OFF_WF   134656     // 32 f32
#define SMEM_BYTES 134784

extern __shared__ __align__(16) char smem_raw[];

// Pack two f32 -> two bf16 (truncate) in ONE v_perm_b32:
// result[15:0] = x[31:16], result[31:16] = y[31:16].
__device__ __forceinline__ unsigned bfpair(float x, float y) {
  return __builtin_amdgcn_perm(__float_as_uint(y), __float_as_uint(x), 0x07060302u);
}
__device__ __forceinline__ unsigned short bf16b(float x) {
  return (unsigned short)(__float_as_uint(x) >> 16);   // -> ds_store_b16_d16_hi pattern
}

// Plain (unpermuted) K index inside a 32-wide chunk so A-fragments are contiguous:
// reader (lane half hr) takes 16 consecutive elems at chunk + hr*16 which map
// to K = hr*16..hr*16+15 (contiguous order, as used for B too).
// writer: h = 16p + l is stored at chunk + h

// One expert stage: A = tile[16 x 32] (expert n slice), B = W[n] (32x32), +bias, relu.
__device__ __forceinline__ void expert_stage(const __bf16* __restrict__ w_lds,
                                             const float* __restrict__ b_lds,
                                             const __bf16* __restrict__ tile,
                                             int l, int half, v8f (&acc)[NUM][2]) {
#pragma unroll
  for (int n = 0; n < NUM; ++n) {
    // A fragment: row = l; contiguous 32B: half takes h = 16*half..16*half+15
    v16bf Af = *(const v16bf*)(tile + l*128 + n*32 + half*16);
#pragma unroll
    for (int j = 0; j < 2; ++j) {
      // B fragment: col k = 16j+l; lanes 0-15 hold h=0..15, lanes 16-31 h=16..31
      v16bf Bf = *(const v16bf*)(w_lds + n*1024 + (j*16 + l)*32 + half*16);
      acc[n][j] = __builtin_amdgcn_wmma_f32_16x16x32_bf16(
          false, Af, false, Bf, (short)0, acc[n][j], false, false);
    }
  }
#pragma unroll
  for (int n = 0; n < NUM; ++n)
#pragma unroll
    for (int j = 0; j < 2; ++j) {
      float bias = b_lds[n*32 + j*16 + l];
#pragma unroll
      for (int v = 0; v < 8; ++v)
        acc[n][j][v] = fmaxf(acc[n][j][v] + bias, 0.0f);
    }
}

// Routing mix: out[b,h,m] = sum_n act[b,h,n]*c[b,n,m];
// write bf16 tile at [row][m*32 + h] (plain contiguous K order).
__device__ __forceinline__ void mix_to_tile(const v8f (&acc)[NUM][2],
                                            const float* __restrict__ c_lds,
                                            unsigned short* __restrict__ tile,
                                            int wave, int l, int half, int phl) {
#pragma unroll
  for (int v = 0; v < 8; ++v) {
    const int rl   = half*8 + v;           // local row in wave tile (0..15)
    const int r128 = wave*16 + rl;         // row within the block's 128 rows
    const float4* c4 = (const float4*)(c_lds + r128*16);
    float4 q0 = c4[0], q1 = c4[1], q2 = c4[2], q3 = c4[3];
    const float ca[16] = {q0.x,q0.y,q0.z,q0.w, q1.x,q1.y,q1.z,q1.w,
                          q2.x,q2.y,q2.z,q2.w, q3.x,q3.y,q3.z,q3.w};
#pragma unroll
    for (int tp = 0; tp < 8; ++tp) {
      const int p = tp & 1, m = tp >> 1;   // h = 16p + l, output expert-slot m
      float f = acc[0][p][v]*ca[0*4+m] + acc[1][p][v]*ca[1*4+m]
              + acc[2][p][v]*ca[2*4+m] + acc[3][p][v]*ca[3*4+m];
      tile[rl*128 + m*32 + p*16 + l] = bf16b(f);
    }
  }
}

__global__ void __launch_bounds__(THREADS)
soft_moe_kernel(const float* __restrict__ xg,
                const float* __restrict__ c1g, const float* __restrict__ c2g,
                const float* __restrict__ c3g,
                const float* __restrict__ W1g, const float* __restrict__ b1g,
                const float* __restrict__ W2g, const float* __restrict__ b2g,
                const float* __restrict__ W3g, const float* __restrict__ b3g,
                const float* __restrict__ Wfg, const float* __restrict__ bfg,
                float* __restrict__ outg) {
  unsigned*       w1_lds = (unsigned*)(smem_raw + OFF_W1);
  unsigned*       w2_lds = (unsigned*)(smem_raw + OFF_W2);
  unsigned*       w3_lds = (unsigned*)(smem_raw + OFF_W3);
  unsigned short* tiles  = (unsigned short*)(smem_raw + OFF_TILE);
  float*          c1_lds = (float*)(smem_raw + OFF_C1);
  float*          c2_lds = (float*)(smem_raw + OFF_C2);
  float*          c3_lds = (float*)(smem_raw + OFF_C3);
  float*          b1_lds = (float*)(smem_raw + OFF_B1);
  float*          b2_lds = (float*)(smem_raw + OFF_B2);
  float*          b3_lds = (float*)(smem_raw + OFF_B3);
  float*          wf_lds = (float*)(smem_raw + OFF_WF);

  const int tid = threadIdx.x;
  const size_t row0 = (size_t)blockIdx.x * ROWS_PER_BLOCK;

  // ---- cooperative preload: weights -> bf16 LDS, routing rows -> f32 LDS ----
  {
    const float2* w1f2 = (const float2*)W1g;        // 16384 bf16 pairs
    for (int i = tid; i < 16384; i += THREADS) { float2 p = w1f2[i]; w1_lds[i] = bfpair(p.x, p.y); }
    const float2* w2f2 = (const float2*)W2g;        // 2048 pairs each
    const float2* w3f2 = (const float2*)W3g;
    for (int i = tid; i < 2048; i += THREADS) { float2 p = w2f2[i]; w2_lds[i] = bfpair(p.x, p.y); }
    for (int i = tid; i < 2048; i += THREADS) { float2 p = w3f2[i]; w3_lds[i] = bfpair(p.x, p.y); }
    for (int i = tid; i < 2048; i += THREADS) c1_lds[i] = c1g[row0*16 + i];
    for (int i = tid; i < 2048; i += THREADS) c2_lds[i] = c2g[row0*16 + i];
    for (int i = tid; i < 512;  i += THREADS) c3_lds[i] = c3g[row0*4 + i];
    if (tid < 128) { b1_lds[tid] = b1g[tid]; b2_lds[tid] = b2g[tid]; b3_lds[tid] = b3g[tid]; }
    if (tid < 32)  wf_lds[tid] = Wfg[tid];
  }
  __syncthreads();

  const int wave = tid >> 5;
  const int lane = tid & 31;
  const int l    = lane & 15;                  // N / A-row index
  const int half = lane >> 4;                  // lane half selects K sub-range
  const int phl  = (l & 7) | ((l >> 3) << 4);  // unused: mix_to_tile stores h = 16p + l unpermuted
  unsigned short* tile_w = tiles + wave * (16*128);
  const __bf16*   tile_r = (const __bf16*)tile_w;
  const __bf16*   w2bf   = (const __bf16*)w2_lds;
  const __bf16*   w3bf   = (const __bf16*)w3_lds;
  const __bf16*   w1bf   = (const __bf16*)w1_lds;

  // ---- stage 1: X[16x256] @ W1^T -> o1[16x128], bias+relu ----
  v8f acc1[NUM][2];
#pragma unroll
  for (int n = 0; n < NUM; ++n) { acc1[n][0] = {}; acc1[n][1] = {}; }

  const float* xrow = xg + (row0 + (size_t)wave*16 + (size_t)l) * IN_DIM;
  __builtin_prefetch(xrow + IN_DIM * 16, 0, 1);   // global_prefetch_b8: warm next tiles
#pragma unroll
  for (int s = 0; s < 8; ++s) {                 // K steps of 32 over IN_DIM
    const float4* xp = (const float4*)(xrow + s*32 + half*16);
    const float4* xq = (const float4*)(xrow + s*32 + half*16 + 8);
    float4 f0 = xp[0], f1 = xp[1], f2 = xq[0], f3 = xq[1];
    v8u au;
    au[0]=bfpair(f0.x,f0.y); au[1]=bfpair(f0.z,f0.w);
    au[2]=bfpair(f1.x,f1.y); au[3]=bfpair(f1.z,f1.w);
    au[4]=bfpair(f2.x,f2.y); au[5]=bfpair(f2.z,f2.w);
    au[6]=bfpair(f3.x,f3.y); au[7]=bfpair(f3.z,f3.w);
    v16bf Af = __builtin_bit_cast(v16bf, au);
#pragma unroll
    for (int t = 0; t < 8; ++t) {               // 8 N tiles of 16 cols
      v16bf Bf = *(const v16bf*)(w1bf + (t*16 + l)*256 + s*32 + half*16);
      acc1[t>>1][t&1] = __builtin_amdgcn_wmma_f32_16x16x32_bf16(
          false, Af, false, Bf, (short)0, acc1[t>>1][t&1], false, false);
    }
  }
#pragma unroll
  for (int n = 0; n < NUM; ++n)
#pragma unroll
    for (int p = 0; p < 2; ++p) {
      float bias = b1_lds[(n*2 + p)*16 + l];
#pragma unroll
      for (int v = 0; v < 8; ++v)
        acc1[n][p][v] = fmaxf(acc1[n][p][v] + bias, 0.0f);
    }

  // ---- mix1 -> in2 tile ----
  mix_to_tile(acc1, c1_lds, tile_w, wave, l, half, phl);

  // ---- stage 2 + mix2 ----
  v8f acc2[NUM][2];
#pragma unroll
  for (int n = 0; n < NUM; ++n) { acc2[n][0] = {}; acc2[n][1] = {}; }
  expert_stage(w2bf, b2_lds, tile_r, l, half, acc2);
  mix_to_tile(acc2, c2_lds, tile_w, wave, l, half, phl);

  // ---- stage 3 ----
  v8f acc3[NUM][2];
#pragma unroll
  for (int n = 0; n < NUM; ++n) { acc3[n][0] = {}; acc3[n][1] = {}; }
  expert_stage(w3bf, b3_lds, tile_r, l, half, acc3);

  // ---- mix3 (c3: [B,NUM,1]) + final Wf dot + cross-lane reduce ----
  const float wfa = wf_lds[l];
  const float wfb = wf_lds[16 + l];
  float red[8];
#pragma unroll
  for (int v = 0; v < 8; ++v) {
    const int rl   = half*8 + v;
    const int r128 = wave*16 + rl;
    const float* cp = c3_lds + r128*4;
    float c0 = cp[0], c1 = cp[1], c2 = cp[2], c3 = cp[3];
    float in4_0 = acc3[0][0][v]*c0 + acc3[1][0][v]*c1 + acc3[2][0][v]*c2 + acc3[3][0][v]*c3;
    float in4_1 = acc3[0][1][v]*c0 + acc3[1][1][v]*c1 + acc3[2][1][v]*c2 + acc3[3][1][v]*c3;
    float sv = wfa*in4_0 + wfb*in4_1;           // partial over h = {l, 16+l}
#pragma unroll
    for (int off = 1; off < 16; off <<= 1)      // reduce over 16 N-lanes per half
      sv += __shfl_xor(sv, off, 32);
    red[v] = sv;
  }
  if (l == 0) {
    const float bfv = bfg[0];
    const size_t rb = row0 + (size_t)wave*16 + (size_t)half*8;
#pragma unroll
    for (int v = 0; v < 8; ++v) outg[rb + v] = red[v] + bfv;
  }
}

extern "C" void kernel_launch(void* const* d_in, const int* in_sizes, int n_in,
                              void* d_out, int out_size, void* d_ws, size_t ws_size,
                              hipStream_t stream) {
  const float* x  = (const float*)d_in[0];
  const float* c1 = (const float*)d_in[1];
  const float* c2 = (const float*)d_in[2];
  const float* c3 = (const float*)d_in[3];
  const float* W1 = (const float*)d_in[4];
  const float* b1 = (const float*)d_in[5];
  const float* W2 = (const float*)d_in[6];
  const float* b2 = (const float*)d_in[7];
  const float* W3 = (const float*)d_in[8];
  const float* b3 = (const float*)d_in[9];
  const float* Wf = (const float*)d_in[10];
  const float* bf = (const float*)d_in[11];
  float* out = (float*)d_out;

  const int Brows = in_sizes[0] / IN_DIM;          // 262144
  const int grid  = Brows / ROWS_PER_BLOCK;        // 2048

  soft_moe_kernel<<<grid, THREADS, SMEM_BYTES, stream>>>(
      x, c1, c2, c3, W1, b1, W2, b2, W3, b3, Wf, bf, out);
}
